// Qwen3VLVisionAttention_31653908971640
// MI455X (gfx1250) — matrix operand order repaired and checked by reading, not compiled, not run on hardware
//
#include <hip/hip_runtime.h>
#include <hip/hip_bf16.h>
#include <stdint.h>

#define S_TOK 16384
#define DIM   1280
#define NHEAD 16
#define SEG   512
#define HD    80
#define HDP   96      // head dim padded to multiple of 32 for K-chunks
#define QKV_N (3*DIM)

typedef __bf16 bf16_t;
typedef __attribute__((ext_vector_type(16))) __bf16 v16bf;
typedef __attribute__((ext_vector_type(8)))  float  v8f;

union FragU { uint4 u[2]; v16bf v; };

__device__ inline v8f zero8() {
    v8f z;
#pragma unroll
    for (int i = 0; i < 8; ++i) z[i] = 0.0f;
    return z;
}

// A fragment (16xK tile, K-chunk of 32 bf16) from a row-major matrix.
// ISA 16-bit A layout: lanes 0-15 row M=lane hold K = {0..7, 16..23};
// lanes 16-31 row M=lane-16 hold K = {8..15, 24..31}.
__device__ inline v16bf load_a_frag(const bf16_t* __restrict__ base, int ld) {
    int lane = threadIdx.x & 31;
    int row  = lane & 15;
    int hi   = lane >> 4;
    const bf16_t* p = base + (size_t)row * ld + hi * 8;
    FragU f;
    f.u[0] = *reinterpret_cast<const uint4*>(p);       // K = hi*8 + 0..7
    f.u[1] = *reinterpret_cast<const uint4*>(p + 16);  // K = 16 + hi*8 + 0..7
    return f.v;
}

// B fragment (Kx16 tile, K-chunk of 32). B[k][n] is supplied as rows of a
// row-major matrix R where R[n][k] = B[k][n] (i.e. column n of B = row n of R).
// 16-bit B layout (matches A): lanes 0-15 col N=lane hold K = {0..7, 16..23}; lanes 16-31 hold K = {8..15, 24..31}.
__device__ inline v16bf load_b_frag(const bf16_t* __restrict__ base, int ld) {
    int lane = threadIdx.x & 31;
    int col  = lane & 15;
    int hi   = lane >> 4;
    const bf16_t* p = base + (size_t)col * ld + hi * 8;
    FragU f;
    f.u[0] = *reinterpret_cast<const uint4*>(p);       // K = hi*8 + 0..7
    f.u[1] = *reinterpret_cast<const uint4*>(p + 16);  // K = 16 + hi*8 + 0..7
    return f.v;
}

// ---------------- fp32 -> bf16 staging ----------------
__global__ void cvt_bf16_kernel(const float* __restrict__ src, bf16_t* __restrict__ dst, int n) {
    int i = blockIdx.x * blockDim.x + threadIdx.x;
    if (i < n) dst[i] = (bf16_t)src[i];
}

// ---------------- QKV GEMM + bias + RoPE + scatter ----------------
// One wave computes 32 tokens x 80 features (one full head slice of q|k|v):
// 2 A fragments share each B fragment -> ~11 FLOP/B of L2 traffic.
__global__ __launch_bounds__(32) void qkv_rope_kernel(
    const bf16_t* __restrict__ hidden,   // [S][DIM]
    const bf16_t* __restrict__ wq,       // [3*DIM][DIM] row-major (= B columns)
    const float*  __restrict__ bias,     // [3*DIM]
    const float*  __restrict__ cosb,     // [S][80]
    const float*  __restrict__ sinb,     // [S][80]
    bf16_t* __restrict__ qbuf,           // [NHEAD][S][HDP]
    bf16_t* __restrict__ kbuf,           // [NHEAD][S][HDP]
    bf16_t* __restrict__ vtbuf)          // [NHEAD][HD][S]
{
    const int m0   = blockIdx.x * 32;    // token tile base (2 x 16)
    const int sl   = blockIdx.y;         // 0..47 head-slice: features sl*80..sl*80+79
    const int sel  = sl / NHEAD;         // 0=q 1=k 2=v
    const int head = sl % NHEAD;
    const int lane = threadIdx.x & 31;
    const int col  = lane & 15;
    const int hi   = lane >> 4;

    v8f acc[2][5];
#pragma unroll
    for (int mi = 0; mi < 2; ++mi)
#pragma unroll
        for (int t = 0; t < 5; ++t) acc[mi][t] = zero8();

    for (int k0 = 0; k0 < DIM; k0 += 32) {
        v16bf a0 = load_a_frag(hidden + (size_t)m0 * DIM + k0, DIM);
        v16bf a1 = load_a_frag(hidden + (size_t)(m0 + 16) * DIM + k0, DIM);
#pragma unroll
        for (int t = 0; t < 5; ++t) {
            int n0 = sl * HD + t * 16;
            v16bf b = load_b_frag(wq + (size_t)n0 * DIM + k0, DIM);
            acc[0][t] = __builtin_amdgcn_wmma_f32_16x16x32_bf16(
                false, a0, false, b, (short)0, acc[0][t], false, false);
            acc[1][t] = __builtin_amdgcn_wmma_f32_16x16x32_bf16(
                false, a1, false, b, (short)0, acc[1][t], false, false);
        }
    }

    // + bias
#pragma unroll
    for (int t = 0; t < 5; ++t) {
        float bv = bias[sl * HD + t * 16 + col];
#pragma unroll
        for (int mi = 0; mi < 2; ++mi)
#pragma unroll
            for (int r = 0; r < 8; ++r) acc[mi][t][r] += bv;
    }

#pragma unroll
    for (int mi = 0; mi < 2; ++mi) {
        const int mb = m0 + mi * 16;
        float res[5][8];
        if (sel < 2) {
            // RoPE: out[d] = x[d]*cos[d] + rot(x)[d]*sin[d],
            // rot(x)[d] = d<40 ? -x[d+40] : x[d-40].  d = t*16 + col; the
            // partner element lives at column col^8 of another tile register.
            const int   srclo[5] = {2, 3, 4, 0, 1};   // source tile when col<8
            const int   srchi[5] = {3, 4, 0, 1, 2};   // source tile when col>=8
            const float sgnlo[5] = {-1.f, -1.f, -1.f, 1.f, 1.f};
            const float sgnhi[5] = {-1.f, -1.f, 1.f, 1.f, 1.f};
#pragma unroll
            for (int t = 0; t < 5; ++t) {
#pragma unroll
                for (int r = 0; r < 8; ++r) {
                    float vlo = __shfl_xor(acc[mi][srclo[t]][r], 8, 32);
                    float vhi = __shfl_xor(acc[mi][srchi[t]][r], 8, 32);
                    float rot = (col & 8) ? sgnhi[t] * vhi : sgnlo[t] * vlo;
                    int tok = mb + r + 8 * hi;
                    int d   = t * 16 + col;
                    float c = cosb[(size_t)tok * HD + d];
                    float s = sinb[(size_t)tok * HD + d];
                    res[t][r] = acc[mi][t][r] * c + rot * s;
                }
            }
        } else {
#pragma unroll
            for (int t = 0; t < 5; ++t)
#pragma unroll
                for (int r = 0; r < 8; ++r) res[t][r] = acc[mi][t][r];
        }

        if (sel < 2) {
            bf16_t* dst = (sel == 0 ? qbuf : kbuf) + (size_t)head * S_TOK * HDP;
#pragma unroll
            for (int t = 0; t < 5; ++t)
#pragma unroll
                for (int r = 0; r < 8; ++r) {
                    int tok = mb + r + 8 * hi;
                    dst[(size_t)tok * HDP + t * 16 + col] = (bf16_t)res[t][r];
                }
            // zero the pad dims 80..95 (read later by the d0=64 K-chunk)
#pragma unroll
            for (int r = 0; r < 8; ++r) {
                int tok = mb + r + 8 * hi;
                dst[(size_t)tok * HDP + HD + col] = (bf16_t)0.0f;
            }
        } else {
            bf16_t* dst = vtbuf + (size_t)head * HD * S_TOK;  // [d][token]
#pragma unroll
            for (int t = 0; t < 5; ++t)
#pragma unroll
                for (int r = 0; r < 8; ++r) {
                    int tok = mb + r + 8 * hi;
                    dst[(size_t)(t * 16 + col) * S_TOK + tok] = (bf16_t)res[t][r];
                }
        }
    }
}

// ---------------- Attention: one wave = 16 query rows of one (seg, head) ----------------
__global__ __launch_bounds__(32) void attn_kernel(
    const bf16_t* __restrict__ qbuf,   // [NHEAD][S][HDP]
    const bf16_t* __restrict__ kbuf,   // [NHEAD][S][HDP]
    const bf16_t* __restrict__ vtbuf,  // [NHEAD][HD][S]
    bf16_t* __restrict__ attn)         // [S][DIM]
{
    __shared__ __align__(16) float  sc[16][SEG];   // scores (q-major)
    __shared__ __align__(16) bf16_t pb[16][SEG];   // probabilities (q-major)
    __shared__ float linv[16];

    const int qt   = blockIdx.x;        // 0..31 query tile within segment
    const int head = blockIdx.y;
    const int seg  = blockIdx.z;
    const int segbase = seg * SEG;
    const int q0   = segbase + qt * 16;
    const int lane = threadIdx.x & 31;
    const int col  = lane & 15;
    const int hi   = lane >> 4;
    const float scaling = 0.1118033988749895f;  // 80^-0.5

    const bf16_t* qh = qbuf + (size_t)head * S_TOK * HDP;
    const bf16_t* kh = kbuf + (size_t)head * S_TOK * HDP;
    const bf16_t* vh = vtbuf + (size_t)head * HD * S_TOK;

    // Phase 1: scores^T tiles (M = k-token, N = q-token) so softmax reduction
    // is along in-lane VGPRs + one shfl_xor(16).
    for (int kt = 0; kt < SEG / 16; ++kt) {
        const int kt0 = segbase + kt * 16;
        v8f s = zero8();
#pragma unroll
        for (int d0 = 0; d0 < HDP; d0 += 32) {
            v16bf a = load_a_frag(kh + (size_t)kt0 * HDP + d0, HDP);  // K rows
            v16bf b = load_b_frag(qh + (size_t)q0 * HDP + d0, HDP);   // Q cols
            s = __builtin_amdgcn_wmma_f32_16x16x32_bf16(
                false, a, false, b, (short)0, s, false, false);
        }
        // lane holds column q=col, rows k = r + 8*hi of this tile
#pragma unroll
        for (int r = 0; r < 8; ++r)
            sc[col][kt * 16 + r + 8 * hi] = s[r] * scaling;
    }
    __syncthreads();

    // Softmax along k for row q=col; lane pair (col, col+16) splits the 512 keys.
    {
        const int row = col;
        float m = -1e30f;
        for (int k = hi * 256; k < hi * 256 + 256; ++k) m = fmaxf(m, sc[row][k]);
        m = fmaxf(m, __shfl_xor(m, 16, 32));
        float ssum = 0.0f;
        for (int k = hi * 256; k < hi * 256 + 256; ++k) {
            float e = __expf(sc[row][k] - m);
            pb[row][k] = (bf16_t)e;
            ssum += e;
        }
        ssum += __shfl_xor(ssum, 16, 32);
        if (lane < 16) linv[lane] = 1.0f / ssum;
    }
    __syncthreads();

    // Phase 2: out(16q x 80d) = P(16 x 512) @ V(512 x 80); P is A from LDS,
    // V^T rows give contiguous B columns.
#pragma unroll
    for (int dt = 0; dt < 5; ++dt) {
        v8f o = zero8();
        for (int k0 = 0; k0 < SEG; k0 += 32) {
            v16bf a = load_a_frag((const bf16_t*)&pb[0][0] + k0, SEG);
            v16bf b = load_b_frag(vh + (size_t)(dt * 16) * S_TOK + segbase + k0, S_TOK);
            o = __builtin_amdgcn_wmma_f32_16x16x32_bf16(
                false, a, false, b, (short)0, o, false, false);
        }
#pragma unroll
        for (int r = 0; r < 8; ++r) {
            int qrow = r + 8 * hi;
            float v = o[r] * linv[qrow];
            attn[(size_t)(q0 + qrow) * DIM + head * HD + dt * 16 + col] = (bf16_t)v;
        }
    }
}

// ---------------- Output projection GEMM ----------------
// One wave computes a 32x64 tile (each B fragment reused by 2 A fragments).
__global__ __launch_bounds__(32) void proj_kernel(
    const bf16_t* __restrict__ attn,  // [S][DIM]
    const bf16_t* __restrict__ wp,    // [DIM][DIM] row-major
    const float*  __restrict__ bias,  // [DIM]
    float* __restrict__ out)          // [S][DIM]
{
    const int m0 = blockIdx.x * 32;
    const int n0 = blockIdx.y * 64;
    const int lane = threadIdx.x & 31;
    const int col  = lane & 15;
    const int hi   = lane >> 4;

    v8f acc[2][4];
#pragma unroll
    for (int mi = 0; mi < 2; ++mi)
#pragma unroll
        for (int t = 0; t < 4; ++t) acc[mi][t] = zero8();

    for (int k0 = 0; k0 < DIM; k0 += 32) {
        v16bf a0 = load_a_frag(attn + (size_t)m0 * DIM + k0, DIM);
        v16bf a1 = load_a_frag(attn + (size_t)(m0 + 16) * DIM + k0, DIM);
#pragma unroll
        for (int t = 0; t < 4; ++t) {
            v16bf b = load_b_frag(wp + (size_t)(n0 + t * 16) * DIM + k0, DIM);
            acc[0][t] = __builtin_amdgcn_wmma_f32_16x16x32_bf16(
                false, a0, false, b, (short)0, acc[0][t], false, false);
            acc[1][t] = __builtin_amdgcn_wmma_f32_16x16x32_bf16(
                false, a1, false, b, (short)0, acc[1][t], false, false);
        }
    }

#pragma unroll
    for (int t = 0; t < 4; ++t) {
        float bv = bias[n0 + t * 16 + col];
#pragma unroll
        for (int mi = 0; mi < 2; ++mi)
#pragma unroll
            for (int r = 0; r < 8; ++r) {
                int tok = m0 + mi * 16 + r + 8 * hi;
                out[(size_t)tok * DIM + n0 + t * 16 + col] = acc[mi][t][r] + bv;
            }
    }
}

extern "C" void kernel_launch(void* const* d_in, const int* in_sizes, int n_in,
                              void* d_out, int out_size, void* d_ws, size_t ws_size,
                              hipStream_t stream) {
    (void)in_sizes; (void)n_in; (void)out_size; (void)ws_size;

    const float* hidden = (const float*)d_in[0];
    // d_in[1] = cu_seqlens (fixed arange(0, S+1, 512) -> segmentation is compile-time)
    const float* cosb   = (const float*)d_in[2];
    const float* sinb   = (const float*)d_in[3];
    const float* qkv_w  = (const float*)d_in[4];
    const float* qkv_b  = (const float*)d_in[5];
    const float* proj_w = (const float*)d_in[6];
    const float* proj_b = (const float*)d_in[7];
    float* out = (float*)d_out;

    // Workspace carve-up (256B aligned)
    char* base = (char*)d_ws;
    size_t off = 0;
    auto carve = [&](size_t bytes) {
        char* p = base + off;
        off = (off + bytes + 255) & ~(size_t)255;
        return p;
    };
    bf16_t* hid_b   = (bf16_t*)carve((size_t)S_TOK * DIM * 2);         // 42 MB
    bf16_t* wqkv_b  = (bf16_t*)carve((size_t)QKV_N * DIM * 2);         // 9.8 MB
    bf16_t* wproj_b = (bf16_t*)carve((size_t)DIM * DIM * 2);           // 3.3 MB
    bf16_t* qbuf    = (bf16_t*)carve((size_t)NHEAD * S_TOK * HDP * 2); // 50 MB
    bf16_t* kbuf    = (bf16_t*)carve((size_t)NHEAD * S_TOK * HDP * 2); // 50 MB
    bf16_t* vtbuf   = (bf16_t*)carve((size_t)NHEAD * HD * S_TOK * 2);  // 42 MB
    bf16_t* attn_b  = (bf16_t*)carve((size_t)S_TOK * DIM * 2);         // 42 MB

    // Stage 0: fp32 -> bf16
    {
        int n1 = S_TOK * DIM;
        cvt_bf16_kernel<<<(n1 + 255) / 256, 256, 0, stream>>>(hidden, hid_b, n1);
        int n2 = QKV_N * DIM;
        cvt_bf16_kernel<<<(n2 + 255) / 256, 256, 0, stream>>>(qkv_w, wqkv_b, n2);
        int n3 = DIM * DIM;
        cvt_bf16_kernel<<<(n3 + 255) / 256, 256, 0, stream>>>(proj_w, wproj_b, n3);
    }

    // Stage 1: QKV GEMM + bias + RoPE, scatter to q/k (row-major, padded) and v^T
    qkv_rope_kernel<<<dim3(S_TOK / 32, 48, 1), 32, 0, stream>>>(
        hid_b, wqkv_b, qkv_b, cosb, sinb, qbuf, kbuf, vtbuf);

    // Stage 2: segmented attention
    attn_kernel<<<dim3(SEG / 16, NHEAD, S_TOK / SEG), 32, 0, stream>>>(
        qbuf, kbuf, vtbuf, attn_b);

    // Stage 3: output projection
    proj_kernel<<<dim3(S_TOK / 32, DIM / 64, 1), 32, 0, stream>>>(
        attn_b, wproj_b, proj_b, out);
}
